// RWKV7cTimeMix_65627100283389
// MI455X (gfx1250) — matrix operand order repaired and checked by reading, not compiled, not run on hardware
//
#include <hip/hip_runtime.h>

typedef __bf16 bf16;
typedef __attribute__((ext_vector_type(16))) __bf16 v16bf;
typedef __attribute__((ext_vector_type(8)))  __bf16 v8bf;
typedef __attribute__((ext_vector_type(8)))  float  v8f;
typedef int v4i __attribute__((vector_size(16)));

#if __has_builtin(__builtin_amdgcn_global_load_async_to_lds_b128)
#define HAVE_ASYNC_LDS 1
typedef __attribute__((address_space(1))) v4i* gptr128_t;
typedef __attribute__((address_space(3))) v4i* lptr128_t;
#else
#define HAVE_ASYNC_LDS 0
#endif

__device__ __forceinline__ void wait_async0() {
#if __has_builtin(__builtin_amdgcn_s_wait_asynccnt)
  __builtin_amdgcn_s_wait_asynccnt(0);
#else
  asm volatile("s_wait_asynccnt 0x0" ::: "memory");
#endif
}

constexpr int Bsz = 4, Tsz = 2048, Csz = 1024, Hsz = 16, Nsz = 64;
constexpr int MTOK = Bsz * Tsz;            // 8192 token rows
constexpr int DMISS = 4096;                // 4*C
constexpr int KMISS = 128;

// ---------------------------------------------------------------------------
// Workspace layout (bytes from base of d_ws)
// ---------------------------------------------------------------------------
constexpr size_t MB = 1024ull * 1024ull;
constexpr size_t OFF_XQ    = 0 * MB;    // bf16 [8192][1024]  16MB
constexpr size_t OFF_XK    = 16 * MB;
constexpr size_t OFF_XV    = 32 * MB;
constexpr size_t OFF_XVF   = 48 * MB;
constexpr size_t OFF_XG    = 64 * MB;   // bf16 [8192][128]    2MB
constexpr size_t OFF_WQT   = 66 * MB;   // bf16 [1024][1024]   2MB
constexpr size_t OFF_WKT   = 68 * MB;
constexpr size_t OFF_WVT   = 70 * MB;
constexpr size_t OFF_WPT   = 72 * MB;
constexpr size_t OFF_MISSB = 74 * MB;   // bf16 [128][4096]    1MB
constexpr size_t OFF_Q     = 80 * MB;   // f32 [8192][1024]   32MB (becomes q4)
constexpr size_t OFF_K     = 112 * MB;  // (becomes k4)
constexpr size_t OFF_V     = 144 * MB;  // (becomes v4)
constexpr size_t OFF_VF    = 176 * MB;
constexpr size_t OFF_DEL   = 208 * MB;  // f32 [8192][4096]  128MB
constexpr size_t OFF_W4    = 336 * MB;  // f32 [8192][1024]
constexpr size_t OFF_Z4    = 368 * MB;
constexpr size_t OFF_B4    = 400 * MB;
constexpr size_t OFF_Y     = 432 * MB;  // f32 scan output
constexpr size_t OFF_YBF   = 464 * MB;  // bf16 final gemm input (ends 480MB)

__device__ __forceinline__ float bfrt(float x) { return (float)(bf16)x; }
__device__ __forceinline__ float sigmoidf_(float x) { return 1.f / (1.f + __expf(-x)); }

// ---------------------------------------------------------------------------
// 1) Token shift + bf16 cast of GEMM A-matrices
// ---------------------------------------------------------------------------
__global__ __launch_bounds__(256) void prep_kernel(
    const float* __restrict__ x, const float* __restrict__ x0,
    const float* __restrict__ dx0,
    const float* __restrict__ mu_q, const float* __restrict__ mu_k,
    const float* __restrict__ mu_v,
    bf16* __restrict__ xq, bf16* __restrict__ xk, bf16* __restrict__ xv,
    bf16* __restrict__ xvf, bf16* __restrict__ xg)
{
  size_t i = (size_t)blockIdx.x * 256 + threadIdx.x;
  if (i >= (size_t)MTOK * Csz) return;
  int c = (int)(i % Csz);
  size_t m = i / Csz;
  int t = (int)(m % Tsz);
  float xc = x[i];
  float prev = (t > 0) ? x[i - Csz] : 0.f;
  float d = prev - xc;                        // dx_prev
  xq[i]  = (bf16)(xc + d * mu_q[c]);
  xk[i]  = (bf16)(xc + d * mu_k[c]);
  xv[i]  = (bf16)(xc + d * mu_v[c]);
  xvf[i] = (bf16)(x0[i] + dx0[i] * mu_v[c]);  // xv_first = x0 + dx0*x_v
  if (c < KMISS) xg[m * KMISS + c] = (bf16)xc;
}

// ---------------------------------------------------------------------------
// 2) Weight transpose-cast:  WT[k][n] = (bf16) W[n][k]
// ---------------------------------------------------------------------------
__global__ __launch_bounds__(256) void cast_transpose_kernel(
    const float* __restrict__ W, bf16* __restrict__ WT, int rows, int cols)
{
  size_t i = (size_t)blockIdx.x * 256 + threadIdx.x;
  if (i >= (size_t)rows * cols) return;
  int n = (int)(i % rows);
  size_t k = i / rows;
  WT[k * rows + n] = (bf16)W[(size_t)n * cols + k];
}

__global__ __launch_bounds__(256) void cast_kernel(
    const float* __restrict__ A, bf16* __restrict__ B, size_t n)
{
  size_t i = (size_t)blockIdx.x * 256 + threadIdx.x;
  if (i < n) B[i] = (bf16)A[i];
}

// ---------------------------------------------------------------------------
// 3) WMMA bf16 GEMM:  C[M][N] = A[M][K] * B[K][N]  (+ optional residual)
//    block = 128 threads (4 waves); tile 128x64, K-step 32.
//    Double-buffered LDS; next tile staged with GLOBAL_LOAD_ASYNC_TO_LDS
//    (ASYNCcnt) when available, else via register double-buffering.
// ---------------------------------------------------------------------------
#define BMt 128
#define BNt 64
#define BKt 32

__global__ __launch_bounds__(128) void gemm_bf16_kernel(
    const bf16* __restrict__ A, int lda,
    const bf16* __restrict__ Bm, int ldb,
    float* __restrict__ C, int ldc,
    const float* __restrict__ resid,
    int K)
{
  __shared__ __attribute__((aligned(16))) bf16 As[2][BMt][40];  // 80B row stride
  __shared__ __attribute__((aligned(16))) bf16 Bs[2][BKt][72];  // 144B row stride

  const int bm = blockIdx.x * BMt;
  const int bn = blockIdx.y * BNt;
  const int tid = threadIdx.x;
  const int wave = tid >> 5;
  const int lane = tid & 31;

  // staging maps: A = 4 passes of 32 rows (4 lanes x 16B per row),
  //               B = 2 passes of 16 rows (8 lanes x 16B per row)
  const int ar = tid >> 2, ac = (tid & 3) * 8;
  const int br = tid >> 3, bc = (tid & 7) * 8;

  v8f acc[2][4];
#pragma unroll
  for (int i = 0; i < 2; ++i)
#pragma unroll
    for (int j = 0; j < 4; ++j) acc[i][j] = v8f{0};

  const int kits = K / BKt;

#if HAVE_ASYNC_LDS
  // prologue: async-stage tile 0 (writes LDS directly, tracked by ASYNCcnt)
#pragma unroll
  for (int p = 0; p < 4; ++p)
    __builtin_amdgcn_global_load_async_to_lds_b128(
        (gptr128_t)(A + (size_t)(bm + p * 32 + ar) * lda + ac),
        (lptr128_t)&As[0][p * 32 + ar][ac], 0, 0);
#pragma unroll
  for (int p = 0; p < 2; ++p)
    __builtin_amdgcn_global_load_async_to_lds_b128(
        (gptr128_t)(Bm + (size_t)(p * 16 + br) * ldb + (bn + bc)),
        (lptr128_t)&Bs[0][p * 16 + br][bc], 0, 0);
  wait_async0();
  __syncthreads();
#else
  v8bf ra[4], rb[2];
#pragma unroll
  for (int p = 0; p < 4; ++p)
    ra[p] = *(const v8bf*)(A + (size_t)(bm + p * 32 + ar) * lda + ac);
#pragma unroll
  for (int p = 0; p < 2; ++p)
    rb[p] = *(const v8bf*)(Bm + (size_t)(p * 16 + br) * ldb + (bn + bc));
#pragma unroll
  for (int p = 0; p < 4; ++p) *(v8bf*)&As[0][p * 32 + ar][ac] = ra[p];
#pragma unroll
  for (int p = 0; p < 2; ++p) *(v8bf*)&Bs[0][p * 16 + br][bc] = rb[p];
  __syncthreads();
#endif

  const int m0  = wave * 32 + (lane & 15);
  const int khh = (lane >> 4) * 8;

  for (int kt = 0; kt < kits; ++kt) {
    const int cur = kt & 1, nxt = cur ^ 1;
    const bool more = (kt + 1 < kits);

    // issue next tile's loads (latency hidden behind WMMA below)
    if (more) {
      const int k0 = (kt + 1) * BKt;
#if HAVE_ASYNC_LDS
#pragma unroll
      for (int p = 0; p < 4; ++p)
        __builtin_amdgcn_global_load_async_to_lds_b128(
            (gptr128_t)(A + (size_t)(bm + p * 32 + ar) * lda + (k0 + ac)),
            (lptr128_t)&As[nxt][p * 32 + ar][ac], 0, 0);
#pragma unroll
      for (int p = 0; p < 2; ++p)
        __builtin_amdgcn_global_load_async_to_lds_b128(
            (gptr128_t)(Bm + (size_t)(k0 + p * 16 + br) * ldb + (bn + bc)),
            (lptr128_t)&Bs[nxt][p * 16 + br][bc], 0, 0);
#else
#pragma unroll
      for (int p = 0; p < 4; ++p)
        ra[p] = *(const v8bf*)(A + (size_t)(bm + p * 32 + ar) * lda + (k0 + ac));
#pragma unroll
      for (int p = 0; p < 2; ++p)
        rb[p] = *(const v8bf*)(Bm + (size_t)(k0 + p * 16 + br) * ldb + (bn + bc));
#endif
    }

    // load all fragments first (single dscnt wait), then stream 8 WMMAs
    v16bf af[2], bfr[4];
#pragma unroll
    for (int i = 0; i < 2; ++i) {
      v8bf lo = *(const v8bf*)&As[cur][m0 + i * 16][khh];
      v8bf hi = *(const v8bf*)&As[cur][m0 + i * 16][16 + khh];
      af[i] = __builtin_shufflevector(lo, hi,
          0,1,2,3,4,5,6,7,8,9,10,11,12,13,14,15);
    }
#pragma unroll
    for (int nt = 0; nt < 4; ++nt) {
#pragma unroll
      for (int s = 0; s < 8; ++s) {
        bfr[nt][s]     = Bs[cur][khh + s][nt * 16 + (lane & 15)];
        bfr[nt][8 + s] = Bs[cur][16 + khh + s][nt * 16 + (lane & 15)];
      }
    }
#pragma unroll
    for (int i = 0; i < 2; ++i)
#pragma unroll
      for (int nt = 0; nt < 4; ++nt)
        acc[i][nt] = __builtin_amdgcn_wmma_f32_16x16x32_bf16(
            false, af[i], false, bfr[nt], (short)0, acc[i][nt], false, false);

    if (more) {
#if HAVE_ASYNC_LDS
      wait_async0();
#else
#pragma unroll
      for (int p = 0; p < 4; ++p) *(v8bf*)&As[nxt][p * 32 + ar][ac] = ra[p];
#pragma unroll
      for (int p = 0; p < 2; ++p) *(v8bf*)&Bs[nxt][p * 16 + br][bc] = rb[p];
#endif
      __syncthreads();
    }
  }

  // D layout: VGPR r -> M = r + 8*(lane>>4), lane&15 -> N
  // Residual null-check hoisted out of the store loops (no per-element branch).
  const int coll = lane & 15;
  if (resid) {
#pragma unroll
    for (int i = 0; i < 2; ++i) {
      const int row0 = bm + wave * 32 + i * 16 + 8 * (lane >> 4);
#pragma unroll
      for (int nt = 0; nt < 4; ++nt) {
        const int col = bn + nt * 16 + coll;
#pragma unroll
        for (int r = 0; r < 8; ++r) {
          size_t off = (size_t)(row0 + r) * ldc + col;
          C[off] = acc[i][nt][r] + resid[off];
        }
      }
    }
  } else {
#pragma unroll
    for (int i = 0; i < 2; ++i) {
      const int row0 = bm + wave * 32 + i * 16 + 8 * (lane >> 4);
#pragma unroll
      for (int nt = 0; nt < 4; ++nt) {
        const int col = bn + nt * 16 + coll;
#pragma unroll
        for (int r = 0; r < 8; ++r) {
          size_t off = (size_t)(row0 + r) * ldc + col;
          C[off] = acc[i][nt][r];
        }
      }
    }
  }
}

// ---------------------------------------------------------------------------
// 4) Per-head elementwise: decay / value-mix / sigmoids / k-normalization,
//    then bf16 round-trip of the six scan inputs. One wave per (token, head).
// ---------------------------------------------------------------------------
__global__ __launch_bounds__(256) void ew2_kernel(
    const float* __restrict__ qin, float* __restrict__ kio,
    float* __restrict__ vio, const float* __restrict__ vfin,
    const float* __restrict__ deltas,
    const float* __restrict__ v0, const float* __restrict__ w0,
    const float* __restrict__ a0,
    float* __restrict__ q4, float* __restrict__ w4,
    float* __restrict__ z4, float* __restrict__ b4)
{
  int gid = blockIdx.x * 8 + (threadIdx.x >> 5);
  int lane = threadIdx.x & 31;
  if (gid >= MTOK * Hsz) return;
  int h = gid % Hsz;
  size_t m = (size_t)(gid / Hsz);
  size_t base = m * Csz + (size_t)h * Nsz;

  float kA = kio[base + lane];
  float kB = kio[base + lane + 32];
  float ss = kA * kA + kB * kB;
#pragma unroll
  for (int o = 16; o > 0; o >>= 1) ss += __shfl_xor(ss, o, 32);
  float inv = 1.f / fmaxf(sqrtf(ss), 1e-12f);

#pragma unroll
  for (int e = 0; e < 2; ++e) {
    int n = lane + 32 * e;
    int c = h * Nsz + n;
    size_t idx = base + n;
    size_t drow = m * (size_t)DMISS;
    float wd = deltas[drow + c];
    float vd = deltas[drow + Csz + c];
    float ad = deltas[drow + 2 * Csz + c];
    // lnw = -softplus(-(w0+wd)) - 0.5 ; log_w = -exp(lnw)
    float nl = -(w0[c] + wd);
    float sp = (nl > 20.f) ? nl : log1pf(__expf(nl));
    float logw = -__expf(-sp - 0.5f);
    float a = sigmoidf_(a0[c] + ad);
    float kv = (e == 0) ? kA : kB;
    float kk = kv * inv;
    float vv = vio[idx];
    vv = vv + (vfin[idx] - vv) * sigmoidf_(v0[c] + vd);
    q4[idx] = bfrt(qin[idx]);
    w4[idx] = bfrt(logw);
    kio[idx] = bfrt(kv * a);   // k4 in place
    vio[idx] = bfrt(vv);       // v4 in place
    z4[idx] = bfrt(-kk);
    b4[idx] = bfrt(kk * a);
  }
}

// ---------------------------------------------------------------------------
// 5) RWKV-7 scan. One block of 64 threads per (b,h); thread = v-row.
//    State row S[64] lives in registers; per-step vectors broadcast via LDS.
//    Next timestep is prefetched (global_prefetch_b8) to hide load latency.
// ---------------------------------------------------------------------------
__global__ __launch_bounds__(64) void scan_kernel(
    const float* __restrict__ q4, const float* __restrict__ w4,
    const float* __restrict__ k4, const float* __restrict__ v4,
    const float* __restrict__ z4, const float* __restrict__ b4,
    float* __restrict__ y)
{
  __shared__ __attribute__((aligned(16))) float qs[Nsz];
  __shared__ __attribute__((aligned(16))) float es[Nsz];
  __shared__ __attribute__((aligned(16))) float ks[Nsz];
  __shared__ __attribute__((aligned(16))) float as[Nsz];
  __shared__ __attribute__((aligned(16))) float bs[Nsz];

  const int bh = blockIdx.x;
  const int b = bh / Hsz, h = bh % Hsz;
  const int row = threadIdx.x;   // value dim 0..63
  float S[Nsz];
#pragma unroll
  for (int j = 0; j < Nsz; ++j) S[j] = 0.f;

  const size_t headoff = (size_t)b * Tsz * Csz + (size_t)h * Nsz;
  for (int t = 0; t < Tsz; ++t) {
    const size_t base = headoff + (size_t)t * Csz;
    qs[row] = q4[base + row];
    es[row] = __expf(w4[base + row]);
    ks[row] = k4[base + row];
    as[row] = z4[base + row];    // a_t = z4 = -kk
    bs[row] = b4[base + row];    // b_t = kk*a
    const float vt = v4[base + row];
    if (t + 1 < Tsz) {
      const size_t nb = base + Csz + row;
      __builtin_prefetch(q4 + nb, 0, 1);
      __builtin_prefetch(w4 + nb, 0, 1);
      __builtin_prefetch(k4 + nb, 0, 1);
      __builtin_prefetch(v4 + nb, 0, 1);
      __builtin_prefetch(z4 + nb, 0, 1);
      __builtin_prefetch(b4 + nb, 0, 1);
    }
    __syncthreads();

    float Sa = 0.f;
#pragma unroll
    for (int j = 0; j < Nsz; ++j) Sa += S[j] * as[j];
    float yv = 0.f;
#pragma unroll
    for (int j = 0; j < Nsz; ++j) {
      S[j] = S[j] * es[j] + (Sa * bs[j] + vt * ks[j]);
      yv += S[j] * qs[j];
    }
    y[base + row] = yv;
    __syncthreads();
  }
}

// ---------------------------------------------------------------------------
// 6) Epilogue elementwise: u-term, gate g, bf16 cast for the output GEMM.
// ---------------------------------------------------------------------------
__global__ __launch_bounds__(256) void ew3_kernel(
    const float* __restrict__ y, const float* __restrict__ v4,
    const float* __restrict__ x, const float* __restrict__ deltas,
    const float* __restrict__ r_k, bf16* __restrict__ ybf)
{
  int gid = blockIdx.x * 8 + (threadIdx.x >> 5);
  int lane = threadIdx.x & 31;
  if (gid >= MTOK * Hsz) return;
  int h = gid % Hsz;
  size_t m = (size_t)(gid / Hsz);
  size_t base = m * Csz + (size_t)h * Nsz;

  float xA = x[base + lane],      xB = x[base + lane + 32];
  float vA = v4[base + lane],     vB = v4[base + lane + 32];
  float sx = xA * xA + xB * xB;
  float sv = vA * vA + vB * vB;
#pragma unroll
  for (int o = 16; o > 0; o >>= 1) {
    sx += __shfl_xor(sx, o, 32);
    sv += __shfl_xor(sv, o, 32);
  }
  float xn = sqrtf(sx);
  float vn = sqrtf(sv);
  float scale = 1.f - xn / (vn + 1e-12f);
  const float inv_sqrtN = 0.125f;  // 64^-0.5

#pragma unroll
  for (int e = 0; e < 2; ++e) {
    int n = lane + 32 * e;
    int c = h * Nsz + n;
    size_t idx = base + n;
    float u = scale * sigmoidf_(r_k[c]);
    float g = 1.f + deltas[m * (size_t)DMISS + 3 * Csz + c];
    float vv = (e == 0) ? vA : vB;
    ybf[idx] = (bf16)((y[idx] * inv_sqrtN + vv * u) * g);
  }
}

// ---------------------------------------------------------------------------
// Launch
// ---------------------------------------------------------------------------
extern "C" void kernel_launch(void* const* d_in, const int* in_sizes, int n_in,
                              void* d_out, int out_size, void* d_ws, size_t ws_size,
                              hipStream_t stream)
{
  const float* residual = (const float*)d_in[0];
  const float* x        = (const float*)d_in[1];
  const float* x0       = (const float*)d_in[2];
  const float* dx0      = (const float*)d_in[3];
  // d_in[4] token_ids: unused
  const float* Wq    = (const float*)d_in[5];
  const float* Wk    = (const float*)d_in[6];
  const float* Wv    = (const float*)d_in[7];
  const float* Wproj = (const float*)d_in[8];
  const float* miss  = (const float*)d_in[9];
  const float* x_q   = (const float*)d_in[10];
  const float* x_k   = (const float*)d_in[11];
  const float* x_v   = (const float*)d_in[12];
  const float* v0    = (const float*)d_in[13];
  const float* w0    = (const float*)d_in[14];
  const float* a0    = (const float*)d_in[15];
  const float* r_k   = (const float*)d_in[16];

  char* ws = (char*)d_ws;
  bf16* xq    = (bf16*)(ws + OFF_XQ);
  bf16* xk    = (bf16*)(ws + OFF_XK);
  bf16* xv    = (bf16*)(ws + OFF_XV);
  bf16* xvf   = (bf16*)(ws + OFF_XVF);
  bf16* xg    = (bf16*)(ws + OFF_XG);
  bf16* WqT   = (bf16*)(ws + OFF_WQT);
  bf16* WkT   = (bf16*)(ws + OFF_WKT);
  bf16* WvT   = (bf16*)(ws + OFF_WVT);
  bf16* WpT   = (bf16*)(ws + OFF_WPT);
  bf16* missB = (bf16*)(ws + OFF_MISSB);
  float* qb   = (float*)(ws + OFF_Q);
  float* kb   = (float*)(ws + OFF_K);
  float* vb   = (float*)(ws + OFF_V);
  float* vfb  = (float*)(ws + OFF_VF);
  float* del  = (float*)(ws + OFF_DEL);
  float* w4   = (float*)(ws + OFF_W4);
  float* z4   = (float*)(ws + OFF_Z4);
  float* b4   = (float*)(ws + OFF_B4);
  float* yb   = (float*)(ws + OFF_Y);
  bf16* ybf   = (bf16*)(ws + OFF_YBF);

  const size_t nElem = (size_t)MTOK * Csz;

  // 1) token shift + casts
  prep_kernel<<<(unsigned)((nElem + 255) / 256), 256, 0, stream>>>(
      x, x0, dx0, x_q, x_k, x_v, xq, xk, xv, xvf, xg);

  // 2) weight transpose-casts
  cast_transpose_kernel<<<(Csz * Csz) / 256, 256, 0, stream>>>(Wq, WqT, Csz, Csz);
  cast_transpose_kernel<<<(Csz * Csz) / 256, 256, 0, stream>>>(Wk, WkT, Csz, Csz);
  cast_transpose_kernel<<<(Csz * Csz) / 256, 256, 0, stream>>>(Wv, WvT, Csz, Csz);
  cast_transpose_kernel<<<(Csz * Csz) / 256, 256, 0, stream>>>(Wproj, WpT, Csz, Csz);
  cast_kernel<<<(KMISS * DMISS) / 256, 256, 0, stream>>>(miss, missB,
                                                         (size_t)KMISS * DMISS);

  // 3) WMMA GEMMs
  dim3 gblk(128);
  dim3 ggrid(MTOK / BMt, Csz / BNt);    // 64 x 16
  gemm_bf16_kernel<<<ggrid, gblk, 0, stream>>>(xq,  Csz, WqT, Csz, qb,  Csz, nullptr, Csz);
  gemm_bf16_kernel<<<ggrid, gblk, 0, stream>>>(xk,  Csz, WkT, Csz, kb,  Csz, nullptr, Csz);
  gemm_bf16_kernel<<<ggrid, gblk, 0, stream>>>(xv,  Csz, WvT, Csz, vb,  Csz, nullptr, Csz);
  gemm_bf16_kernel<<<ggrid, gblk, 0, stream>>>(xvf, Csz, WvT, Csz, vfb, Csz, nullptr, Csz);
  dim3 mgrid(MTOK / BMt, DMISS / BNt);  // 64 x 64
  gemm_bf16_kernel<<<mgrid, gblk, 0, stream>>>(xg, KMISS, missB, DMISS, del, DMISS,
                                               nullptr, KMISS);

  // 4) per-head elementwise -> scan inputs (q4 in qb, k4 in kb, v4 in vb)
  ew2_kernel<<<(MTOK * Hsz) / 8, 256, 0, stream>>>(
      qb, kb, vb, vfb, del, v0, w0, a0, qb, w4, z4, b4);

  // 5) recurrence
  scan_kernel<<<Bsz * Hsz, 64, 0, stream>>>(qb, w4, kb, vb, z4, b4, yb);

  // 6) epilogue elementwise
  ew3_kernel<<<(MTOK * Hsz) / 8, 256, 0, stream>>>(yb, vb, x, del, r_k, ybf);

  // 7) output projection + residual
  gemm_bf16_kernel<<<ggrid, gblk, 0, stream>>>(ybf, Csz, WpT, Csz, (float*)d_out, Csz,
                                               residual, Csz);
}
